// QLSTM_65481071403909
// MI455X (gfx1250) — compile-verified, matrix operand order checked by reading, not run on hardware
//
#include <hip/hip_runtime.h>

#define T_STEPS 512
#define BSZ 256
#define DIN 128
#define HID 256
#define NP 128
#define DH 384   // DIN + HID
#define NG 1024  // 4*HID

typedef __attribute__((ext_vector_type(16))) _Float16 v16h;
typedef __attribute__((ext_vector_type(8)))  float    v8f;

union Frag16 { v16h v; uint4 q[2]; };

// Load a 16-bit A/B WMMA fragment from a row-major [row][k] array.
// Per ISA layout (16-bit 16x32): lane half hi=0 holds K={0..7, 16..23},
// hi=1 holds K={8..15, 24..31} -> two contiguous 16B reads.
__device__ __forceinline__ v16h load_frag(const _Float16* base, int hi) {
    Frag16 f;
    f.q[0] = *(const uint4*)(base + hi * 8);
    f.q[1] = *(const uint4*)(base + 16 + hi * 8);
    return f.v;
}

__device__ __forceinline__ float fast_sig(float x) {
    return __builtin_amdgcn_rcpf(1.0f + __expf(-x));
}
__device__ __forceinline__ float fast_tanh(float x) {
    return 1.0f - 2.0f * __builtin_amdgcn_rcpf(__expf(2.0f * x) + 1.0f);
}

// One-time: convert prototypes and stacked gate weights to f16, compute |p|^2.
__global__ void qlstm_prep(const float* __restrict__ protos,
                           const float* __restrict__ Wf, const float* __restrict__ Wi,
                           const float* __restrict__ Wg, const float* __restrict__ Wo,
                           _Float16* __restrict__ protoT,   // [NP][DH]
                           _Float16* __restrict__ WcatT,    // [NG][NP], n = g*HID + h
                           float* __restrict__ pnorm) {
    int i = blockIdx.x * blockDim.x + threadIdx.x;
    if (i < NP * DH) protoT[i] = (_Float16)protos[i];
    if (i < NG * NP) {
        int n = i >> 7, kk = i & 127;
        int g = n >> 8, h = n & 255;
        const float* W = (g == 0) ? Wf : (g == 1) ? Wi : (g == 2) ? Wg : Wo;
        WcatT[i] = (_Float16)W[h * NP + kk];
    }
    if (i < NP) {
        float s = 0.f;
        for (int j = 0; j < DH; ++j) { float v = protos[i * DH + j]; s += v * v; }
        pnorm[i] = s;
    }
}

// Persistent recurrent kernel: block = 16 batch rows, 8 waves, full T loop.
__launch_bounds__(256)
__global__ void qlstm_main(const float* __restrict__ x,          // [T,B,D]
                           const _Float16* __restrict__ protoT,  // [NP][DH]
                           const _Float16* __restrict__ WcatT,   // [NG][NP]
                           const float* __restrict__ pnorm,
                           const float* __restrict__ bf, const float* __restrict__ bi,
                           const float* __restrict__ bg, const float* __restrict__ bo,
                           float* __restrict__ out,              // [T,B,H]
                           float* __restrict__ hx_out, float* __restrict__ cx_out) {
    __shared__ _Float16 s_comb[16 * DH];   // [m][k] combined = [x | hx] (f16)
    __shared__ _Float16 s_k[16 * NP];      // [m][p] RBF kernel values (f16)
    __shared__ float    s_cnorm[16];

    const int tid  = threadIdx.x;
    const int wv   = tid >> 5;     // wave 0..7
    const int lane = tid & 31;
    const int n16  = lane & 15;
    const int hi   = lane >> 4;
    const int b0   = blockIdx.x * 16;
    const int mrow = tid >> 4;     // row this thread services for x/cnorm
    const int chk  = tid & 15;     // 16 threads (one half-wave) per row

    // Per-thread t-invariant constants.
    const float pn = pnorm[wv * 16 + n16];
    float bfv[2], biv[2], bgv[2], bov[2];
#pragma unroll
    for (int ht = 0; ht < 2; ++ht) {
        int h = wv * 32 + ht * 16 + n16;
        bfv[ht] = bf[h]; biv[ht] = bi[h]; bgv[ht] = bg[h]; bov[ht] = bo[h];
    }

    // Prototype B-fragments: pin 10 of 12 in VGPRs; the last two are streamed
    // from L2 each step (issued pre-barrier, hidden under the dot GEMM).
    const _Float16* brow = protoT + (size_t)(wv * 16 + n16) * DH;
    v16h pb[10];
#pragma unroll
    for (int kc = 0; kc < 10; ++kc) pb[kc] = load_frag(brow + kc * 32, hi);

    // Gate-weight fragment addressing: wave owns h in [32w, 32w+32) for 4 gates.
    const _Float16* wbase = WcatT + (size_t)(wv * 32 + n16) * NP;

    // Cell state in registers: (m = r + 8*hi, h = 32*wv + 16*ht + n16)
    float cxr[2][8] = {};

    // hx = 0 initially -> zero the hidden part of s_comb
    {
        int kk = DIN + chk * 16;
        uint4 z; z.x = z.y = z.z = z.w = 0u;
        *(uint4*)(&s_comb[mrow * DH + kk])     = z;
        *(uint4*)(&s_comb[mrow * DH + kk + 8]) = z;
    }

    // Register double-buffer of this thread's 8 x-values; preload t=0.
    float4 xa, xb;
    {
        const float* src = x + ((size_t)0 * BSZ + b0 + mrow) * DIN + chk * 8;
        xa = *(const float4*)src;
        xb = *(const float4*)(src + 4);
    }
    __syncthreads();

    for (int t = 0; t < T_STEPS; ++t) {
        // 1) convert x[t] tile -> s_comb[:, 0:DIN]; start cnorm with own x-part
        union { _Float16 h[8]; uint4 q; } cv;
        cv.h[0] = (_Float16)xa.x; cv.h[1] = (_Float16)xa.y;
        cv.h[2] = (_Float16)xa.z; cv.h[3] = (_Float16)xa.w;
        cv.h[4] = (_Float16)xb.x; cv.h[5] = (_Float16)xb.y;
        cv.h[6] = (_Float16)xb.z; cv.h[7] = (_Float16)xb.w;
        *(uint4*)(&s_comb[mrow * DH + chk * 8]) = cv.q;

        float part = 0.f;
#pragma unroll
        for (int j = 0; j < 8; ++j) { float v = (float)cv.h[j]; part += v * v; }
        {   // h-part of |combined|^2 (valid: written before last barrier)
            const _Float16* hp = &s_comb[mrow * DH + DIN + chk * 16];
#pragma unroll
            for (int j = 0; j < 16; ++j) { float v = (float)hp[j]; part += v * v; }
        }
        part += __shfl_xor(part, 1);
        part += __shfl_xor(part, 2);
        part += __shfl_xor(part, 4);
        part += __shfl_xor(part, 8);
        if (chk == 0) s_cnorm[mrow] = part;

        // issue next step's x loads now; they complete during the GEMMs below
        if (t + 1 < T_STEPS) {
            const float* nx = x + ((size_t)(t + 1) * BSZ + b0 + mrow) * DIN + chk * 8;
            xa = *(const float4*)nx;
            xb = *(const float4*)(nx + 4);
        }
        if (t + 2 < T_STEPS)
            __builtin_prefetch(x + ((size_t)(t + 2) * BSZ + b0 + mrow) * DIN + chk * 8, 0, 3);

        // Pre-barrier L2 loads (depend only on constants): gate-weight chunk 0
        // and the two unpinned prototype fragments. Latency hides under B1+dot.
        v16h wbuf[8];
#pragma unroll
        for (int g = 0; g < 4; ++g)
#pragma unroll
            for (int ht = 0; ht < 2; ++ht)
                wbuf[g * 2 + ht] = load_frag(wbase + ((size_t)g * HID + ht * 16) * NP, hi);
        v16h pbA = load_frag(brow + 10 * 32, hi);
        v16h pbB = load_frag(brow + 11 * 32, hi);

        __syncthreads();   // B1: x tile + cnorm visible

        // 2) dot = combined @ protos^T : [16,384]x[384,128]; wave owns 16 protos.
        //    Two accumulators + paired loads break the serial WMMA/wait chain.
        v8f acc0 = {}, acc1 = {};
        const _Float16* arow = &s_comb[n16 * DH];
#pragma unroll
        for (int kc = 0; kc < 10; kc += 2) {
            v16h a0 = load_frag(arow + kc * 32, hi);
            v16h a1 = load_frag(arow + (kc + 1) * 32, hi);
            acc0 = __builtin_amdgcn_wmma_f32_16x16x32_f16(false, a0, false, pb[kc],
                                                          (short)0, acc0, false, false);
            acc1 = __builtin_amdgcn_wmma_f32_16x16x32_f16(false, a1, false, pb[kc + 1],
                                                          (short)0, acc1, false, false);
        }
        {
            v16h a0 = load_frag(arow + 10 * 32, hi);
            v16h a1 = load_frag(arow + 11 * 32, hi);
            acc0 = __builtin_amdgcn_wmma_f32_16x16x32_f16(false, a0, false, pbA,
                                                          (short)0, acc0, false, false);
            acc1 = __builtin_amdgcn_wmma_f32_16x16x32_f16(false, a1, false, pbB,
                                                          (short)0, acc1, false, false);
        }

        // 3) k = exp(-(|c|^2 + |p|^2 - 2 dot)) -> s_k (A operand of gates GEMM)
#pragma unroll
        for (int r = 0; r < 8; ++r) {
            int m = r + 8 * hi;
            float d2 = s_cnorm[m] + pn - 2.0f * (acc0[r] + acc1[r]);
            s_k[m * NP + (wv * 16 + n16)] = (_Float16)__expf(-d2);
        }
        __syncthreads();   // B2: s_k ready

        // 4) gates = k @ Wcat : [16,128]x[128,1024]; 8 independent accumulators.
        //    Rolling-pair A-fragments (2 in flight) + single-buffered B chunks
        //    (chunk kc+1 loads issue after chunk kc's WMMAs -> max WAR distance).
        const _Float16* karow = &s_k[n16 * NP];
        v16h kaA = load_frag(karow, hi);
        v16h kaB = load_frag(karow + 32, hi);

        v8f ga[4][2] = {};
#pragma unroll
        for (int kc = 0; kc < 4; ++kc) {
            v16h acur = (kc & 1) ? kaB : kaA;
#pragma unroll
            for (int g = 0; g < 4; ++g)
#pragma unroll
                for (int ht = 0; ht < 2; ++ht)
                    ga[g][ht] = __builtin_amdgcn_wmma_f32_16x16x32_f16(
                        false, acur, false, wbuf[g * 2 + ht],
                        (short)0, ga[g][ht], false, false);
            if (kc + 2 < 4) {   // refill the register pair just consumed
                if (kc & 1) kaB = load_frag(karow + (kc + 2) * 32, hi);
                else        kaA = load_frag(karow + (kc + 2) * 32, hi);
            }
            if (kc < 3) {
#pragma unroll
                for (int g = 0; g < 4; ++g)
#pragma unroll
                    for (int ht = 0; ht < 2; ++ht)
                        wbuf[g * 2 + ht] =
                            load_frag(wbase + ((size_t)g * HID + ht * 16) * NP + (kc + 1) * 32, hi);
            }
        }

        // 5) pointwise LSTM; cx in registers; hx -> LDS (next step) + HBM output
#pragma unroll
        for (int ht = 0; ht < 2; ++ht) {
            int h = wv * 32 + ht * 16 + n16;
#pragma unroll
            for (int r = 0; r < 8; ++r) {
                int m = r + 8 * hi;
                float fv = fast_sig(ga[0][ht][r] + bfv[ht]);
                float iv = fast_sig(ga[1][ht][r] + biv[ht]);
                float gv = fast_tanh(ga[2][ht][r] + bgv[ht]);
                float ov = fast_sig(ga[3][ht][r] + bov[ht]);
                float c  = fv * cxr[ht][r] + iv * gv;
                cxr[ht][r] = c;
                float hv = ov * fast_tanh(c);
                s_comb[m * DH + DIN + h] = (_Float16)hv;
                out[((size_t)t * BSZ + b0 + m) * HID + h] = hv;
                if (t == T_STEPS - 1) {
                    hx_out[(size_t)(b0 + m) * HID + h] = hv;
                    cx_out[(size_t)(b0 + m) * HID + h] = c;
                }
            }
        }
        __syncthreads();   // B3: end of step (protects s_comb hx + s_k reuse)
    }
}

extern "C" void kernel_launch(void* const* d_in, const int* in_sizes, int n_in,
                              void* d_out, int out_size, void* d_ws, size_t ws_size,
                              hipStream_t stream) {
    const float* inputs = (const float*)d_in[0];
    const float* protos = (const float*)d_in[1];
    const float* Wf = (const float*)d_in[2];
    const float* bf = (const float*)d_in[3];
    const float* Wi = (const float*)d_in[4];
    const float* bi = (const float*)d_in[5];
    const float* Wg = (const float*)d_in[6];
    const float* bg = (const float*)d_in[7];
    const float* Wo = (const float*)d_in[8];
    const float* bo = (const float*)d_in[9];

    char* ws = (char*)d_ws;
    _Float16* protoT = (_Float16*)ws;                      // 98304 B
    _Float16* WcatT  = (_Float16*)(ws + 98304);            // 262144 B
    float*    pnorm  = (float*)(ws + 98304 + 262144);      // 512 B

    float* out    = (float*)d_out;
    float* hx_out = out + (size_t)T_STEPS * BSZ * HID;
    float* cx_out = hx_out + (size_t)BSZ * HID;

    qlstm_prep<<<512, 256, 0, stream>>>(protos, Wf, Wi, Wg, Wo, protoT, WcatT, pnorm);
    qlstm_main<<<BSZ / 16, 256, 0, stream>>>(inputs, protoT, WcatT, pnorm,
                                             bf, bi, bg, bo, out, hx_out, cx_out);
}
